// Recon_78683800862964
// MI455X (gfx1250) — compile-verified, matrix operand order checked by reading, not run on hardware
//
#include <hip/hip_runtime.h>
#include <hip/hip_bf16.h>
#include <math.h>

// ---------------------------------------------------------------------------
// MI455X / gfx1250 implementation.
// GEMM-like work (Mamba projections, 3x3 convs as implicit GEMM) goes through
// v_wmma_f32_16x16x32_bf16 (bf16 operands, f32 accumulate). Sequential /
// pointwise work (LayerNorm, depthwise conv1d+SiLU, selective scan, BN, CCE
// gates, final conv+sigmoid) stays fp32 VALU. wave32 everywhere.
// ---------------------------------------------------------------------------

typedef __attribute__((ext_vector_type(16))) __bf16 v16bf;
typedef __attribute__((ext_vector_type(8)))  float  v8f;

union Frag16 { uint4 u[2]; v16bf v; };

__device__ __forceinline__ unsigned short f2bf(float f) {
    unsigned u = __float_as_uint(f);
    u += 0x7fffu + ((u >> 16) & 1u);          // round-to-nearest-even
    return (unsigned short)(u >> 16);
}
__device__ __forceinline__ float sigf(float x)      { return 1.f / (1.f + __expf(-x)); }
__device__ __forceinline__ float softplusf(float x) { return x > 20.f ? x : log1pf(__expf(x)); }

// ---------------------------------------------------------------------------
// LayerNorm over last dim (128), fp32 in -> bf16 out. One wave32 per row.
// ---------------------------------------------------------------------------
__global__ void __launch_bounds__(128) ln_bf16_kernel(
    const float* __restrict__ x, const float* __restrict__ g,
    const float* __restrict__ b, unsigned short* __restrict__ out, int rows)
{
    int wave = threadIdx.x >> 5, lane = threadIdx.x & 31;
    int row = blockIdx.x * 4 + wave;
    if (row >= rows) return;
    const float* xr = x + (size_t)row * 128;
    float v0 = xr[lane], v1 = xr[lane + 32], v2 = xr[lane + 64], v3 = xr[lane + 96];
    float s  = v0 + v1 + v2 + v3;
    float s2 = v0 * v0 + v1 * v1 + v2 * v2 + v3 * v3;
    #pragma unroll
    for (int m = 16; m >= 1; m >>= 1) {
        s  += __shfl_xor(s,  m, 32);
        s2 += __shfl_xor(s2, m, 32);
    }
    float mean = s * (1.f / 128.f);
    float var  = s2 * (1.f / 128.f) - mean * mean;
    float rstd = rsqrtf(var + 1e-5f);
    unsigned short* o = out + (size_t)row * 128;
    o[lane]      = f2bf((v0 - mean) * rstd * g[lane]      + b[lane]);
    o[lane + 32] = f2bf((v1 - mean) * rstd * g[lane + 32] + b[lane + 32]);
    o[lane + 64] = f2bf((v2 - mean) * rstd * g[lane + 64] + b[lane + 64]);
    o[lane + 96] = f2bf((v3 - mean) * rstd * g[lane + 96] + b[lane + 96]);
}

// ---------------------------------------------------------------------------
// Generic WMMA GEMM: C[M,N] = A[M,K](bf16) * W[N,K](bf16)^T (+ Res).
// 8 waves per block; each wave computes a 16x16 C tile. K step = 32.
// Fragment layout per CDNA5 ISA (16-bit A 16x32):
//   lanes 0-15 : M=lane,    K in {0..7} U {16..23}
//   lanes 16-31: M=lane-16, K in {8..15} U {24..31}
// B mirrors with N in place of M.
// ---------------------------------------------------------------------------
__global__ void __launch_bounds__(256) wmma_gemm_bf16(
    const unsigned short* __restrict__ A, const unsigned short* __restrict__ W,
    const float* __restrict__ Res, float* __restrict__ C, int M, int N, int K)
{
    int wave = threadIdx.x >> 5, lane = threadIdx.x & 31;
    int m0 = blockIdx.y * 64 + (wave & 3) * 16;
    int n0 = blockIdx.x * 32 + (wave >> 2) * 16;
    if (m0 >= M || n0 >= N) return;
    int ml  = lane & 15;
    int klo = (lane >> 4) ? 8 : 0;
    const uint4* Arow = (const uint4*)(A + (size_t)(m0 + ml) * K);
    const uint4* Brow = (const uint4*)(W + (size_t)(n0 + ml) * K);
    v8f acc = {};
    for (int k0 = 0; k0 < K; k0 += 32) {
        Frag16 af, bfr;
        int c0 = (k0 + klo) >> 3;              // uint4 = 8 bf16
        af.u[0]  = Arow[c0];  af.u[1]  = Arow[c0 + 2];
        bfr.u[0] = Brow[c0];  bfr.u[1] = Brow[c0 + 2];
        acc = __builtin_amdgcn_wmma_f32_16x16x32_bf16(
            false, af.v, false, bfr.v, (short)0, acc, false, false);
    }
    int mrow = m0 + ((lane >> 4) << 3);
    int ncol = n0 + ml;
    #pragma unroll
    for (int r = 0; r < 8; ++r) {
        size_t idx = (size_t)(mrow + r) * N + ncol;
        float v = acc[r];
        if (Res) v += Res[idx];
        C[idx] = v;
    }
}

// ---------------------------------------------------------------------------
// Causal depthwise conv1d (kernel 4) + SiLU. xz layout (B,L,512); xm = first
// 256 channels. Emits fp32 (scan input u) and bf16 (xproj GEMM A).
// ---------------------------------------------------------------------------
__global__ void __launch_bounds__(256) dwconv_silu_kernel(
    const float* __restrict__ xz, const float* __restrict__ cw,
    const float* __restrict__ cb, float* __restrict__ xm,
    unsigned short* __restrict__ xmb)
{
    int idx = blockIdx.x * 256 + threadIdx.x;          // (b*1024 + l)*256 + d
    if (idx >= 16 * 1024 * 256) return;
    int d = idx & 255, l = (idx >> 8) & 1023, b = idx >> 18;
    float acc = cb[d];
    #pragma unroll
    for (int k = 0; k < 4; ++k) {
        int ls = l - 3 + k;
        if (ls >= 0) acc += xz[(((size_t)b * 1024 + ls) * 512) + d] * cw[d * 4 + k];
    }
    float sv = acc * sigf(acc);
    xm[idx]  = sv;
    xmb[idx] = f2bf(sv);
}

// ---------------------------------------------------------------------------
// Selective scan. One block per batch element; thread = channel d (256).
// dt/B/C row (24 floats from xdbl, N padded to 32) broadcast through LDS.
// Fuses softplus(dt), D-skip and SiLU(z) gate; outputs bf16 for out_proj.
// ---------------------------------------------------------------------------
__global__ void __launch_bounds__(256) scan_kernel(
    const float* __restrict__ xm, const float* __restrict__ xz,
    const float* __restrict__ xdbl, const float* __restrict__ dtw_g,
    const float* __restrict__ dtb_g, const float* __restrict__ Alog,
    const float* __restrict__ Dp, unsigned short* __restrict__ yg)
{
    int b = blockIdx.x, d = threadIdx.x;
    float a[8], dtw[8], h[8];
    #pragma unroll
    for (int n = 0; n < 8; ++n) {
        a[n]   = -__expf(Alog[d * 8 + n]);
        dtw[n] = dtw_g[d * 8 + n];
        h[n]   = 0.f;
    }
    float dtb = dtb_g[d], Dd = Dp[d];
    __shared__ float row[24];                  // [dt_r 8][B 8][C 8]
    for (int l = 0; l < 1024; ++l) {
        __syncthreads();
        if (d < 24) row[d] = xdbl[((size_t)(b * 1024 + l) * 32) + d];
        __syncthreads();
        float u = xm[((size_t)(b * 1024 + l) << 8) + d];
        float dtr = dtb;
        #pragma unroll
        for (int r = 0; r < 8; ++r) dtr += row[r] * dtw[r];
        float dt = softplusf(dtr);
        float y = 0.f;
        #pragma unroll
        for (int n = 0; n < 8; ++n) {
            float dA = __expf(dt * a[n]);
            h[n] = dA * h[n] + dt * u * row[8 + n];
            y += h[n] * row[16 + n];
        }
        y += u * Dd;
        float z = xz[((size_t)(b * 1024 + l) * 512) + 256 + d];
        y *= z * sigf(z);
        yg[((size_t)(b * 1024 + l) << 8) + d] = f2bf(y);
    }
}

// ---------------------------------------------------------------------------
// Conv c1 as implicit-GEMM WMMA: 128 -> 64 ch, 3x3 SAME on 64x64, with the
// 2x upsample fused into A-gather (source is LN output, (16,1024,128) bf16).
// Epilogue: +bias, BN, ReLU -> bf16 (b,h,w,o) for c2.
// ---------------------------------------------------------------------------
__global__ void __launch_bounds__(256) conv1_wmma_kernel(
    const unsigned short* __restrict__ X, const unsigned short* __restrict__ Wp,
    const float* __restrict__ bias, const float* __restrict__ bng,
    const float* __restrict__ bnb, const float* __restrict__ bnm,
    const float* __restrict__ bnv, unsigned short* __restrict__ Y)
{
    int wave = threadIdx.x >> 5, lane = threadIdx.x & 31;
    int p0 = blockIdx.y * 64 + (wave & 3) * 16;
    int n0 = blockIdx.x * 32 + (wave >> 2) * 16;
    int ml = lane & 15, klo = (lane >> 4) ? 8 : 0;
    int p = p0 + ml;                                // output pixel of A-row
    int b = p >> 12, hw = p & 4095, h = hw >> 6, w = hw & 63;
    const uint4* Brow = (const uint4*)(Wp + (size_t)(n0 + ml) * 1152);
    v8f acc = {};
    for (int tap = 0; tap < 9; ++tap) {
        int kh = tap / 3, kw = tap % 3;
        int ih = h + kh - 1, iw = w + kw - 1;
        bool valid = ((unsigned)ih < 64u) && ((unsigned)iw < 64u);
        const uint4* Arow = valid
            ? (const uint4*)(X + (((size_t)b * 1024) + ((ih >> 1) * 32 + (iw >> 1))) * 128)
            : nullptr;
        for (int kc = 0; kc < 128; kc += 32) {
            Frag16 af = {}, bfr;
            if (valid) {
                int ca = (kc + klo) >> 3;
                af.u[0] = Arow[ca]; af.u[1] = Arow[ca + 2];
            }
            int cb = (tap * 128 + kc + klo) >> 3;
            bfr.u[0] = Brow[cb]; bfr.u[1] = Brow[cb + 2];
            acc = __builtin_amdgcn_wmma_f32_16x16x32_bf16(
                false, af.v, false, bfr.v, (short)0, acc, false, false);
        }
    }
    int mrow = p0 + ((lane >> 4) << 3);
    int o = n0 + ml;
    float bscale = rsqrtf(bnv[o] + 1e-5f) * bng[o];
    #pragma unroll
    for (int r = 0; r < 8; ++r) {
        float v = acc[r] + bias[o];
        v = (v - bnm[o]) * bscale + bnb[o];
        v = fmaxf(v, 0.f);
        Y[(size_t)(mrow + r) * 64 + o] = f2bf(v);
    }
}

// ---------------------------------------------------------------------------
// Conv c2 implicit-GEMM WMMA: 64 -> 32 ch, 3x3 SAME on 128x128, upsample
// fused (source (16,64,64,64) bf16 channel-last). Epilogue: +bias, ReLU,
// then BN -> fp32 (b,h,w,c) "combined" buffer.
// ---------------------------------------------------------------------------
__global__ void __launch_bounds__(256) conv2_wmma_kernel(
    const unsigned short* __restrict__ X, const unsigned short* __restrict__ Wp,
    const float* __restrict__ bias, const float* __restrict__ bng,
    const float* __restrict__ bnb, const float* __restrict__ bnm,
    const float* __restrict__ bnv, float* __restrict__ Y)
{
    int wave = threadIdx.x >> 5, lane = threadIdx.x & 31;
    int p0 = blockIdx.y * 64 + (wave & 3) * 16;
    int n0 = (wave >> 2) * 16;                      // N = 32 total
    int ml = lane & 15, klo = (lane >> 4) ? 8 : 0;
    int p = p0 + ml;
    int b = p >> 14, hw = p & 16383, h = hw >> 7, w = hw & 127;
    const uint4* Brow = (const uint4*)(Wp + (size_t)(n0 + ml) * 576);
    v8f acc = {};
    for (int tap = 0; tap < 9; ++tap) {
        int kh = tap / 3, kw = tap % 3;
        int ih = h + kh - 1, iw = w + kw - 1;
        bool valid = ((unsigned)ih < 128u) && ((unsigned)iw < 128u);
        const uint4* Arow = valid
            ? (const uint4*)(X + (((size_t)b * 4096) + ((ih >> 1) * 64 + (iw >> 1))) * 64)
            : nullptr;
        for (int kc = 0; kc < 64; kc += 32) {
            Frag16 af = {}, bfr;
            if (valid) {
                int ca = (kc + klo) >> 3;
                af.u[0] = Arow[ca]; af.u[1] = Arow[ca + 2];
            }
            int cb = (tap * 64 + kc + klo) >> 3;
            bfr.u[0] = Brow[cb]; bfr.u[1] = Brow[cb + 2];
            acc = __builtin_amdgcn_wmma_f32_16x16x32_bf16(
                false, af.v, false, bfr.v, (short)0, acc, false, false);
        }
    }
    int mrow = p0 + ((lane >> 4) << 3);
    int o = n0 + ml;
    float bscale = rsqrtf(bnv[o] + 1e-5f) * bng[o];
    #pragma unroll
    for (int r = 0; r < 8; ++r) {
        float v = fmaxf(acc[r] + bias[o], 0.f);     // relu BEFORE bn (per ref)
        v = (v - bnm[o]) * bscale + bnb[o];
        Y[(size_t)(mrow + r) * 32 + o] = v;
    }
}

// ---------------------------------------------------------------------------
// CCE gate: per (b,c) compute sigmoid(max_hw(a) + mean_hw(a)),
// a = 1x1conv(X, W1). One block per (b,c).
// ---------------------------------------------------------------------------
__global__ void __launch_bounds__(256) cce_gate_kernel(
    const float* __restrict__ X, const float* __restrict__ W1,
    float* __restrict__ gate)
{
    int b = blockIdx.x >> 5, c = blockIdx.x & 31, tid = threadIdx.x;
    float wr[32];
    #pragma unroll
    for (int k = 0; k < 32; ++k) wr[k] = W1[c * 32 + k];
    float mx = -3.402823466e38f, sm = 0.f;
    const float* Xb = X + (size_t)b * 32 * 16384;
    for (int p = tid; p < 16384; p += 256) {
        float a = 0.f;
        #pragma unroll
        for (int k = 0; k < 32; ++k) a += Xb[(size_t)k * 16384 + p] * wr[k];
        mx = fmaxf(mx, a);
        sm += a;
    }
    __shared__ float smax[256], ssum[256];
    smax[tid] = mx; ssum[tid] = sm;
    __syncthreads();
    for (int s = 128; s > 0; s >>= 1) {
        if (tid < s) {
            smax[tid] = fmaxf(smax[tid], smax[tid + s]);
            ssum[tid] += ssum[tid + s];
        }
        __syncthreads();
    }
    if (tid == 0) gate[blockIdx.x] = sigf(smax[0] + ssum[0] * (1.f / 16384.f));
}

// ---------------------------------------------------------------------------
// Add both CCE branches into combined: comb[b,p,c] +=
//   conv1x1(orig_y,W2)*gate_xy + conv1x1(orig_x,W2)*gate_yx
// ---------------------------------------------------------------------------
__global__ void __launch_bounds__(256) cce_combine_kernel(
    const float* __restrict__ ox, const float* __restrict__ oy,
    const float* __restrict__ W2, const float* __restrict__ gxy,
    const float* __restrict__ gyx, float* __restrict__ comb)
{
    int idx = blockIdx.x * 256 + threadIdx.x;       // b*16384 + p
    if (idx >= 16 * 16384) return;
    int b = idx >> 14, p = idx & 16383;
    float xv[32], yv[32];
    #pragma unroll
    for (int k = 0; k < 32; ++k) {
        xv[k] = ox[(((size_t)b * 32 + k) << 14) + p];
        yv[k] = oy[(((size_t)b * 32 + k) << 14) + p];
    }
    float* cr = comb + (size_t)idx * 32;
    for (int c = 0; c < 32; ++c) {
        float r1 = 0.f, r2 = 0.f;
        #pragma unroll
        for (int k = 0; k < 32; ++k) {
            float w = W2[c * 32 + k];
            r1 += yv[k] * w;
            r2 += xv[k] * w;
        }
        cr[c] += r1 * gxy[b * 32 + c] + r2 * gyx[b * 32 + c];
    }
}

// ---------------------------------------------------------------------------
// Final 3x3 conv 32->1 + sigmoid.
// ---------------------------------------------------------------------------
__global__ void __launch_bounds__(256) conv3_sig_kernel(
    const float* __restrict__ comb, const float* __restrict__ W3,
    const float* __restrict__ b3, float* __restrict__ out)
{
    int idx = blockIdx.x * 256 + threadIdx.x;
    if (idx >= 16 * 16384) return;
    int b = idx >> 14, hw = idx & 16383, h = hw >> 7, w = hw & 127;
    float acc = b3[0];
    for (int kh = 0; kh < 3; ++kh) {
        int ih = h + kh - 1;
        if ((unsigned)ih >= 128u) continue;
        for (int kw = 0; kw < 3; ++kw) {
            int iw = w + kw - 1;
            if ((unsigned)iw >= 128u) continue;
            const float* cr = comb + (size_t)((b << 14) + (ih << 7) + iw) * 32;
            int tap = kh * 3 + kw;
            #pragma unroll
            for (int c = 0; c < 32; ++c) acc += cr[c] * W3[c * 9 + tap];
        }
    }
    out[idx] = sigf(acc);
}

// ------------------------- weight prep kernels -----------------------------
__global__ void f2bf_kernel(const float* __restrict__ s,
                            unsigned short* __restrict__ d, int n)
{
    int i = blockIdx.x * 256 + threadIdx.x;
    if (i < n) d[i] = f2bf(s[i]);
}
// (O,I,3,3) -> bf16 (O, 9, I)  [K ordered (tap, ci), contiguous per out-ch]
__global__ void pack_conv_kernel(const float* __restrict__ w,
                                 unsigned short* __restrict__ d, int O, int I)
{
    int idx = blockIdx.x * 256 + threadIdx.x;
    if (idx >= O * 9 * I) return;
    int i = idx % I, tap = (idx / I) % 9, o = idx / (9 * I);
    d[((size_t)o * 9 + tap) * I + i] = f2bf(w[((size_t)o * I + i) * 9 + tap]);
}
// (3,24,256) -> bf16 (3,32,256) zero-padded rows 24..31
__global__ void pad_xproj_kernel(const float* __restrict__ w,
                                 unsigned short* __restrict__ d)
{
    int idx = blockIdx.x * 256 + threadIdx.x;
    if (idx >= 3 * 32 * 256) return;
    int k = idx & 255, n = (idx >> 8) & 31, blk = idx >> 13;
    d[idx] = f2bf(n < 24 ? w[((size_t)blk * 24 + n) * 256 + k] : 0.f);
}

// ---------------------------------------------------------------------------
extern "C" void kernel_launch(void* const* d_in, const int* in_sizes, int n_in,
                              void* d_out, int out_size, void* d_ws, size_t ws_size,
                              hipStream_t stream)
{
    const float* fused0  = (const float*)d_in[0];
    const float* orig_x  = (const float*)d_in[1];
    const float* orig_y  = (const float*)d_in[2];
    const float* m_ln_g  = (const float*)d_in[3];
    const float* m_ln_b  = (const float*)d_in[4];
    const float* m_in_w  = (const float*)d_in[5];
    const float* m_cw    = (const float*)d_in[6];
    const float* m_cb    = (const float*)d_in[7];
    const float* m_xpw   = (const float*)d_in[8];
    const float* m_dtw   = (const float*)d_in[9];
    const float* m_dtb   = (const float*)d_in[10];
    const float* m_Alog  = (const float*)d_in[11];
    const float* m_D     = (const float*)d_in[12];
    const float* m_ow    = (const float*)d_in[13];
    const float* lnf_g   = (const float*)d_in[14];
    const float* lnf_b   = (const float*)d_in[15];
    const float* c1_w    = (const float*)d_in[16];
    const float* c1_b    = (const float*)d_in[17];
    const float* bn1_g   = (const float*)d_in[18];
    const float* bn1_b   = (const float*)d_in[19];
    const float* bn1_m   = (const float*)d_in[20];
    const float* bn1_v   = (const float*)d_in[21];
    const float* c2_w    = (const float*)d_in[22];
    const float* c2_b    = (const float*)d_in[23];
    const float* bn2_g   = (const float*)d_in[24];
    const float* bn2_b   = (const float*)d_in[25];
    const float* bn2_m   = (const float*)d_in[26];
    const float* bn2_v   = (const float*)d_in[27];
    const float* cce_w1  = (const float*)d_in[28];
    const float* cce_w2  = (const float*)d_in[29];
    const float* c3_w    = (const float*)d_in[30];
    const float* c3_b    = (const float*)d_in[31];

    // ---- workspace layout (~83 MB, with reuse) ----
    char* ws = (char*)d_ws;
    size_t off = 0;
    auto alloc = [&](size_t bytes) {
        size_t o = off; off += (bytes + 255) & ~(size_t)255; return o;
    };
    float*  xz      = (float*)(ws + alloc(33554432));  // (16,1024,512); reused as comb
    float*  comb    = xz;                              // (16,128,128,32) in decoder phase
    float*  xm      = (float*)(ws + alloc(16777216));  // (16,1024,256)
    unsigned short* xmb   = (unsigned short*)(ws + alloc(8388608)); // bf16; reused conv1_out
    unsigned short* c1out = xmb;
    float*  xdbl    = (float*)(ws + alloc(2097152));   // (16384,32)
    unsigned short* yg    = (unsigned short*)(ws + alloc(8388608)); // (16384,256) bf16
    float*  fusedw  = (float*)(ws + alloc(8388608));   // (16,1024,128)
    unsigned short* hbf   = (unsigned short*)(ws + alloc(4194304)); // LN out bf16
    float*  gates   = (float*)(ws + alloc(4096));      // g_xy[512], g_yx[512]
    float*  g_xy = gates, *g_yx = gates + 512;
    unsigned short* w_in  = (unsigned short*)(ws + alloc(786432));  // (3,512,128) bf16
    unsigned short* w_xp  = (unsigned short*)(ws + alloc(49152));   // (3,32,256) bf16 padded
    unsigned short* w_out = (unsigned short*)(ws + alloc(196608));  // (3,128,256) bf16
    unsigned short* w_c1  = (unsigned short*)(ws + alloc(147456));  // (64, 9*128) bf16
    unsigned short* w_c2  = (unsigned short*)(ws + alloc(36864));   // (32, 9*64) bf16

    // ---- weight prep (bf16 convert / pad / pack) ----
    f2bf_kernel<<<(196608 + 255) / 256, 256, 0, stream>>>(m_in_w, w_in, 196608);
    f2bf_kernel<<<(98304 + 255) / 256, 256, 0, stream>>>(m_ow, w_out, 98304);
    pad_xproj_kernel<<<(24576 + 255) / 256, 256, 0, stream>>>(m_xpw, w_xp);
    pack_conv_kernel<<<(64 * 9 * 128 + 255) / 256, 256, 0, stream>>>(c1_w, w_c1, 64, 128);
    pack_conv_kernel<<<(32 * 9 * 64 + 255) / 256, 256, 0, stream>>>(c2_w, w_c2, 32, 64);

    const int M = 16384;                // 16 * 1024 tokens
    // ---- 3 pre-LN Mamba blocks with residual ----
    for (int i = 0; i < 3; ++i) {
        const float* src = (i == 0) ? fused0 : fusedw;
        ln_bf16_kernel<<<4096, 128, 0, stream>>>(
            src, m_ln_g + i * 128, m_ln_b + i * 128, hbf, M);
        // in_proj: (16384,128) x (512,128)^T -> xz
        wmma_gemm_bf16<<<dim3(512 / 32, M / 64), 256, 0, stream>>>(
            hbf, w_in + (size_t)i * 512 * 128, nullptr, xz, M, 512, 128);
        // depthwise causal conv1d + SiLU
        dwconv_silu_kernel<<<(16 * 1024 * 256) / 256, 256, 0, stream>>>(
            xz, m_cw + i * 256 * 4, m_cb + i * 256, xm, xmb);
        // x_proj: (16384,256) x (32,256)^T -> xdbl (24 valid cols)
        wmma_gemm_bf16<<<dim3(1, M / 64), 256, 0, stream>>>(
            xmb, w_xp + (size_t)i * 32 * 256, nullptr, xdbl, M, 32, 256);
        // selective scan + dt-proj + softplus + D-skip + SiLU(z) gate
        scan_kernel<<<16, 256, 0, stream>>>(
            xm, xz, xdbl, m_dtw + i * 256 * 8, m_dtb + i * 256,
            m_Alog + i * 256 * 8, m_D + i * 256, yg);
        // out_proj + residual (in-place safe: 1:1 read/write per thread)
        wmma_gemm_bf16<<<dim3(128 / 32, M / 64), 256, 0, stream>>>(
            yg, w_out + (size_t)i * 128 * 256, src, fusedw, M, 128, 256);
    }

    // ---- final LN -> bf16 (feeds conv1 implicit GEMM) ----
    ln_bf16_kernel<<<4096, 128, 0, stream>>>(fusedw, lnf_g, lnf_b, hbf, M);

    // ---- decoder: up2 + c1(+BN+ReLU), up2 + c2(+ReLU+BN) via WMMA ----
    conv1_wmma_kernel<<<dim3(2, 65536 / 64), 256, 0, stream>>>(
        hbf, w_c1, c1_b, bn1_g, bn1_b, bn1_m, bn1_v, c1out);
    conv2_wmma_kernel<<<dim3(1, 262144 / 64), 256, 0, stream>>>(
        c1out, w_c2, c2_b, bn2_g, bn2_b, bn2_m, bn2_v, comb);

    // ---- CCE gates + gated 1x1 contributions ----
    cce_gate_kernel<<<512, 256, 0, stream>>>(orig_x, cce_w1, g_xy);
    cce_gate_kernel<<<512, 256, 0, stream>>>(orig_y, cce_w1, g_yx);
    cce_combine_kernel<<<(16 * 16384) / 256, 256, 0, stream>>>(
        orig_x, orig_y, cce_w2, g_xy, g_yx, comb);

    // ---- final 3x3 conv + sigmoid -> d_out (16,1,128,128) fp32 ----
    conv3_sig_kernel<<<(16 * 16384) / 256, 256, 0, stream>>>(
        comb, c3_w, c3_b, (float*)d_out);
}
